// FusionModel_5385888989234
// MI455X (gfx1250) — matrix operand order repaired and checked by reading, not compiled, not run on hardware
//
#include <hip/hip_runtime.h>
#include <hip/hip_bf16.h>
#include <math.h>

typedef __attribute__((ext_vector_type(16))) _Float16 v16h;
typedef __attribute__((ext_vector_type(8)))  _Float16 v8h;
typedef __attribute__((ext_vector_type(8)))  float    v8f;

#define BSZ   8192
#define DNIR  215
#define DLIBS 244
#define EPS   1e-5f

__device__ __forceinline__ v16h combine16(v8h lo, v8h hi) {
  v16h r;
#pragma unroll
  for (int i = 0; i < 8; ++i) { r[i] = lo[i]; r[i + 8] = hi[i]; }
  return r;
}

// DPP16 lane-permute (pure VALU, co-executes with XDL WMMA; stays within 16-lane rows)
template <int CTRL>
__device__ __forceinline__ float dppmov(float x) {
  int xi = __float_as_int(x);
  return __int_as_float(__builtin_amdgcn_update_dpp(xi, xi, CTRL, 0xF, 0xF, true));
}
// 16-lane butterfly: xor1 (quad_perm), xor2 (quad_perm), half-mirror, mirror
__device__ __forceinline__ float rowmax16(float x) {
  x = fmaxf(x, dppmov<0xB1>(x));   // quad_perm [1,0,3,2]
  x = fmaxf(x, dppmov<0x4E>(x));   // quad_perm [2,3,0,1]
  x = fmaxf(x, dppmov<0x141>(x));  // row_half_mirror
  x = fmaxf(x, dppmov<0x140>(x));  // row_mirror
  return x;
}
__device__ __forceinline__ float rowsum16(float x) {
  x += dppmov<0xB1>(x);
  x += dppmov<0x4E>(x);
  x += dppmov<0x141>(x);
  x += dppmov<0x140>(x);
  return x;
}

// ---------------- Feature extractor: LN(Din) -> Linear(Din,64) -> ReLU ----------------
__global__ __launch_bounds__(64) void extract_kernel(
    const float* __restrict__ x, const float* __restrict__ g, const float* __restrict__ b,
    const float* __restrict__ W, const float* __restrict__ bias,
    float* __restrict__ out, int Din)
{
  __shared__ float xn[256];
  __shared__ float r1[64], r2[64];
  const int row = blockIdx.x, tid = threadIdx.x;
  const float* xr = x + (size_t)row * Din;
  float s = 0.f, s2 = 0.f;
  for (int k = tid; k < Din; k += 64) { float v = xr[k]; s += v; s2 += v * v; }
  r1[tid] = s; r2[tid] = s2;
  __syncthreads();
  float mean = 0.f, msq = 0.f;
  for (int k = 0; k < 64; ++k) { mean += r1[k]; msq += r2[k]; }
  mean /= (float)Din; msq /= (float)Din;
  float rstd = rsqrtf(msq - mean * mean + EPS);
  for (int k = tid; k < Din; k += 64) xn[k] = (xr[k] - mean) * rstd * g[k] + b[k];
  __syncthreads();
  float acc = bias[tid];
  for (int k = 0; k < Din; ++k) acc = fmaf(xn[k], W[k * 64 + tid], acc);
  out[(size_t)row * 64 + tid] = fmaxf(acc, 0.f);
}

// ---------------- 64x64 projection, f32 in -> f16 out (optionally pre-scaled) ----------
__global__ __launch_bounds__(256) void proj_kernel(
    const float* __restrict__ in, const float* __restrict__ W, const float* __restrict__ bias,
    _Float16* __restrict__ out, float scale)
{
  __shared__ float xin[4][64];
  const int tid = threadIdx.x;
  const int lr = tid >> 6, col = tid & 63;
  const int row = blockIdx.x * 4 + lr;
  xin[lr][col] = in[(size_t)row * 64 + col];
  __syncthreads();
  float acc = bias[col];
#pragma unroll 8
  for (int k = 0; k < 64; ++k) acc = fmaf(xin[lr][k], W[k * 64 + col], acc);
  out[(size_t)row * 64 + col] = (_Float16)(acc * scale);
}

// ---------------- Flash attention over the batch axis, WMMA f32_16x16x32_f16 ----------
// 8 waves/WG, each wave owns a 16-query tile; WG iterates shared 32-key blocks with
// double-buffered LDS staging. Q is pre-scaled by (1/sqrt(64))*log2(e): exp2 domain.
__global__ __launch_bounds__(256) void flash_kernel(
    const _Float16* __restrict__ Q0, const _Float16* __restrict__ K0,
    const _Float16* __restrict__ V0, float* __restrict__ O0,
    const _Float16* __restrict__ Q1, const _Float16* __restrict__ K1,
    const _Float16* __restrict__ V1, float* __restrict__ O1)
{
  constexpr int KST = 72;   // K tile row stride (halves): 144B, 16B aligned, conflict pad
  constexpr int VST = 40;   // Vt / P row stride (halves): 80B, 16B aligned
  __shared__ alignas(16) _Float16 Klds[2][32 * KST];   // 32 keys x 64 feats, x2 buffers
  __shared__ alignas(16) _Float16 Vtlds[2][64 * VST];  // 64 feats x 32 keys (transposed)
  __shared__ alignas(16) _Float16 Plds[8 * 16 * VST];  // per-wave 16x32 probabilities

  const _Float16* Q = blockIdx.y ? Q1 : Q0;
  const _Float16* K = blockIdx.y ? K1 : K0;
  const _Float16* V = blockIdx.y ? V1 : V0;
  float*          O = blockIdx.y ? O1 : O0;

  const int tid  = threadIdx.x;
  const int w    = tid >> 5, lane = tid & 31;
  const int g    = lane >> 4, n = lane & 15;
  const int qbase = blockIdx.x * 128 + w * 16;
  const int trow = tid >> 3;          // staging: key row 0..31
  const int tcol = (tid & 7) << 3;    // staging: feature col 0,8,..,56

  // Q A-fragments (row m = lane%16; halves: K = 16*(h/8) + 8*g + h%8 per chunk)
  v16h qa[2];
#pragma unroll
  for (int c = 0; c < 2; ++c) {
    const _Float16* p = Q + (size_t)(qbase + n) * 64 + c * 32 + g * 8;
    qa[c] = combine16(*(const v8h*)p, *(const v8h*)(p + 16));
  }

  float mrow[8], lrow[8];
  v8f of[4] = {};
#pragma unroll
  for (int r = 0; r < 8; ++r) { mrow[r] = -1e30f; lrow[r] = 0.f; }

  _Float16* myP = &Plds[w * 16 * VST];

  // prologue: stage tile 0
  v8h kv = *(const v8h*)(K + (size_t)trow * 64 + tcol);
  v8h vv = *(const v8h*)(V + (size_t)trow * 64 + tcol);
  *(v8h*)(&Klds[0][trow * KST + tcol]) = kv;
#pragma unroll
  for (int i = 0; i < 8; ++i) Vtlds[0][(tcol + i) * VST + trow] = vv[i];

  for (int kb = 0; kb < BSZ; kb += 32) {
    const int cur = (kb >> 5) & 1;
    const bool more = (kb + 32) < BSZ;
    if (more) {   // issue next tile's global loads early; latency hides under WMMA
      kv = *(const v8h*)(K + (size_t)(kb + 32 + trow) * 64 + tcol);
      vv = *(const v8h*)(V + (size_t)(kb + 32 + trow) * 64 + tcol);
    }
    __syncthreads();   // tile[cur] staged; previous compute drained

    // S[16 x 32] = Q * K^T  (two 16-key halves, K-dim 64 = 2 x k32)
    v8f s[2];
#pragma unroll
    for (int kh = 0; kh < 2; ++kh) {
      v8f acc = {};
#pragma unroll
      for (int c = 0; c < 2; ++c) {
        const _Float16* kp = &Klds[cur][(kh * 16 + n) * KST + c * 32 + g * 8];
        v16h bf = combine16(*(const v8h*)kp, *(const v8h*)(kp + 16));
        acc = __builtin_amdgcn_wmma_f32_16x16x32_f16(false, qa[c], false, bf,
                                                     (short)0, acc, false, false);
      }
      s[kh] = acc;
    }

    // online softmax in exp2 domain (row m = r + 8g lives in one 16-lane half)
#pragma unroll
    for (int r = 0; r < 8; ++r) {
      float v0 = s[0][r], v1 = s[1][r];
      float mx = rowmax16(fmaxf(v0, v1));
      float mnew  = fmaxf(mrow[r], mx);
      float alpha = exp2f(mrow[r] - mnew);
      float p0 = exp2f(v0 - mnew);
      float p1 = exp2f(v1 - mnew);
      float rs = rowsum16(p0 + p1);
      lrow[r] = lrow[r] * alpha + rs;
      mrow[r] = mnew;
#pragma unroll
      for (int f = 0; f < 4; ++f) of[f][r] *= alpha;
      _Float16* prow = myP + (r + 8 * g) * VST;   // C-layout -> LDS (row-major 16x32)
      prow[n]      = (_Float16)p0;
      prow[16 + n] = (_Float16)p1;
    }

    // P back as A-fragment (wave-local LDS; DS ops are in-order per wave)
    v16h pa;
    {
      const _Float16* pb = myP + n * VST + 8 * g;
      pa = combine16(*(const v8h*)pb, *(const v8h*)(pb + 16));
    }
    // O[16 x 64] += P[16 x 32] * V[32 x 64]
#pragma unroll
    for (int f = 0; f < 4; ++f) {
      const _Float16* vp = &Vtlds[cur][(f * 16 + n) * VST + g * 8];
      v16h vb = combine16(*(const v8h*)vp, *(const v8h*)(vp + 16));
      of[f] = __builtin_amdgcn_wmma_f32_16x16x32_f16(false, pa, false, vb,
                                                     (short)0, of[f], false, false);
    }

    if (more) {   // drain prefetched tile into the alternate buffer
      *(v8h*)(&Klds[cur ^ 1][trow * KST + tcol]) = kv;
#pragma unroll
      for (int i = 0; i < 8; ++i) Vtlds[cur ^ 1][(tcol + i) * VST + trow] = vv[i];
    }
  }

#pragma unroll
  for (int r = 0; r < 8; ++r) {
    float inv = 1.0f / lrow[r];
    float* orow = O + (size_t)(qbase + r + 8 * g) * 64 + n;
#pragma unroll
    for (int f = 0; f < 4; ++f) orow[f * 16] = of[f][r] * inv;
  }
}

// ---------------- Residual blocks + gate + head, one row per 64-thread block ----------
__global__ __launch_bounds__(64) void head_kernel(
    const float* __restrict__ attn_n2l, const float* __restrict__ attn_l2n,
    const float* __restrict__ nirF, const float* __restrict__ libsF,
    const float* ln1_g, const float* ln1_b, const float* fc1_w, const float* fc1_b,
    const float* ln2_g, const float* ln2_b, const float* fc2_w, const float* fc2_b,
    const float* ln0_g, const float* ln0_b, const float* gate_w, const float* gate_b,
    const float* ln3_g, const float* ln3_b, const float* fc3_w, const float* fc3_b,
    const float* ln4_g, const float* ln4_b, const float* fc4_w, const float* fc4_b,
    float* __restrict__ out)
{
  __shared__ float sa[64], sn[64], sc[128], scn[128], sh[32];
  const int j = threadIdx.x;
  const size_t row = blockIdx.x;

  auto ln_stats = [](const float* arr, int nvals, float& mean, float& rstd) {
    float s = 0.f, s2 = 0.f;
    for (int k = 0; k < nvals; ++k) { float v = arr[k]; s += v; s2 += v * v; }
    mean = s / nvals;
    rstd = rsqrtf(s2 / nvals - mean * mean + EPS);
  };

  float mean, rstd;
  // out1 = relu(LN(attn_nir_to_libs)@fc1 + b1) + libs
  float a1 = attn_n2l[row * 64 + j];
  sa[j] = a1; __syncthreads();
  ln_stats(sa, 64, mean, rstd);
  sn[j] = (a1 - mean) * rstd * ln1_g[j] + ln1_b[j];
  __syncthreads();
  float acc = fc1_b[j];
  for (int k = 0; k < 64; ++k) acc = fmaf(sn[k], fc1_w[k * 64 + j], acc);
  float o1 = fmaxf(acc, 0.f) + libsF[row * 64 + j];
  __syncthreads();

  // out2 = relu(LN(attn_libs_to_nir)@fc2 + b2) + nir
  float a2 = attn_l2n[row * 64 + j];
  sa[j] = a2; __syncthreads();
  ln_stats(sa, 64, mean, rstd);
  sn[j] = (a2 - mean) * rstd * ln2_g[j] + ln2_b[j];
  __syncthreads();
  acc = fc2_b[j];
  for (int k = 0; k < 64; ++k) acc = fmaf(sn[k], fc2_w[k * 64 + j], acc);
  float o2 = fmaxf(acc, 0.f) + nirF[row * 64 + j];

  // sigmoid gate fusion
  sc[j] = o1; sc[64 + j] = o2; __syncthreads();
  ln_stats(sc, 128, mean, rstd);
  scn[j]      = (o1 - mean) * rstd * ln0_g[j]      + ln0_b[j];
  scn[64 + j] = (o2 - mean) * rstd * ln0_g[64 + j] + ln0_b[64 + j];
  __syncthreads();
  acc = gate_b[j];
  for (int k = 0; k < 128; ++k) acc = fmaf(scn[k], gate_w[k * 64 + j], acc);
  float gate = 1.f / (1.f + expf(-acc));
  float fu = gate * o1 + (1.f - gate) * o2;
  __syncthreads();

  // head: LN -> Linear(64,32) -> ReLU -> LN -> Linear(32,1)
  sa[j] = fu; __syncthreads();
  ln_stats(sa, 64, mean, rstd);
  sn[j] = (fu - mean) * rstd * ln3_g[j] + ln3_b[j];
  __syncthreads();
  if (j < 32) {
    float h = fc3_b[j];
    for (int k = 0; k < 64; ++k) h = fmaf(sn[k], fc3_w[k * 32 + j], h);
    sh[j] = fmaxf(h, 0.f);
  }
  __syncthreads();
  if (j == 0) {
    float m2, r2v; ln_stats(sh, 32, m2, r2v);
    float o = fc4_b[0];
    for (int k = 0; k < 32; ++k)
      o = fmaf((sh[k] - m2) * r2v * ln4_g[k] + ln4_b[k], fc4_w[k], o);
    out[row] = o;
  }
}

extern "C" void kernel_launch(void* const* d_in, const int* in_sizes, int n_in,
                              void* d_out, int out_size, void* d_ws, size_t ws_size,
                              hipStream_t stream)
{
  (void)in_sizes; (void)n_in; (void)out_size; (void)ws_size;
  const float* nir_data  = (const float*)d_in[0];
  const float* libs_data = (const float*)d_in[1];
  const float* nir_ln_g  = (const float*)d_in[2];
  const float* nir_ln_b  = (const float*)d_in[3];
  const float* nir_w     = (const float*)d_in[4];
  const float* nir_b     = (const float*)d_in[5];
  const float* libs_ln_g = (const float*)d_in[6];
  const float* libs_ln_b = (const float*)d_in[7];
  const float* libs_w    = (const float*)d_in[8];
  const float* libs_b    = (const float*)d_in[9];
  const float* q_ln_w = (const float*)d_in[10]; const float* q_ln_b = (const float*)d_in[11];
  const float* k_ln_w = (const float*)d_in[12]; const float* k_ln_b = (const float*)d_in[13];
  const float* v_ln_w = (const float*)d_in[14]; const float* v_ln_b = (const float*)d_in[15];
  const float* q_nl_w = (const float*)d_in[16]; const float* q_nl_b = (const float*)d_in[17];
  const float* k_nl_w = (const float*)d_in[18]; const float* k_nl_b = (const float*)d_in[19];
  const float* v_nl_w = (const float*)d_in[20]; const float* v_nl_b = (const float*)d_in[21];
  const float* ln1_g = (const float*)d_in[22]; const float* ln1_b = (const float*)d_in[23];
  const float* fc1_w = (const float*)d_in[24]; const float* fc1_b = (const float*)d_in[25];
  const float* ln2_g = (const float*)d_in[26]; const float* ln2_b = (const float*)d_in[27];
  const float* fc2_w = (const float*)d_in[28]; const float* fc2_b = (const float*)d_in[29];
  const float* ln0_g = (const float*)d_in[30]; const float* ln0_b = (const float*)d_in[31];
  const float* gate_w = (const float*)d_in[32]; const float* gate_b = (const float*)d_in[33];
  const float* ln3_g = (const float*)d_in[34]; const float* ln3_b = (const float*)d_in[35];
  const float* fc3_w = (const float*)d_in[36]; const float* fc3_b = (const float*)d_in[37];
  const float* ln4_g = (const float*)d_in[38]; const float* ln4_b = (const float*)d_in[39];
  const float* fc4_w = (const float*)d_in[40]; const float* fc4_b = (const float*)d_in[41];

  float* ws = (float*)d_ws;
  const size_t NB = (size_t)BSZ * 64;
  float* nirF      = ws;
  float* libsF     = ws + NB;
  float* attn_l2n  = ws + 2 * NB;
  float* attn_n2l  = ws + 3 * NB;
  _Float16* h16 = (_Float16*)(ws + 4 * NB);
  _Float16* Qlibs = h16;            _Float16* Knir  = h16 + NB;
  _Float16* Vnir  = h16 + 2 * NB;   _Float16* Qnir  = h16 + 3 * NB;
  _Float16* Klibs = h16 + 4 * NB;   _Float16* Vlibs = h16 + 5 * NB;

  const float SC = 0.125f * 1.44269504088896f;  // (1/sqrt(64)) * log2(e), folded into Q

  extract_kernel<<<BSZ, 64, 0, stream>>>(nir_data,  nir_ln_g,  nir_ln_b,  nir_w,  nir_b,  nirF,  DNIR);
  extract_kernel<<<BSZ, 64, 0, stream>>>(libs_data, libs_ln_g, libs_ln_b, libs_w, libs_b, libsF, DLIBS);

  proj_kernel<<<BSZ / 4, 256, 0, stream>>>(libsF, q_ln_w, q_ln_b, Qlibs, SC);
  proj_kernel<<<BSZ / 4, 256, 0, stream>>>(nirF,  k_ln_w, k_ln_b, Knir,  1.0f);
  proj_kernel<<<BSZ / 4, 256, 0, stream>>>(nirF,  v_ln_w, v_ln_b, Vnir,  1.0f);
  proj_kernel<<<BSZ / 4, 256, 0, stream>>>(nirF,  q_nl_w, q_nl_b, Qnir,  SC);
  proj_kernel<<<BSZ / 4, 256, 0, stream>>>(libsF, k_nl_w, k_nl_b, Klibs, 1.0f);
  proj_kernel<<<BSZ / 4, 256, 0, stream>>>(libsF, v_nl_w, v_nl_b, Vlibs, 1.0f);

  flash_kernel<<<dim3(BSZ / 128, 2), 256, 0, stream>>>(
      Qlibs, Knir, Vnir, attn_l2n,      // attn_libs_to_nir
      Qnir,  Klibs, Vlibs, attn_n2l);   // attn_nir_to_libs

  head_kernel<<<BSZ, 64, 0, stream>>>(attn_n2l, attn_l2n, nirF, libsF,
      ln1_g, ln1_b, fc1_w, fc1_b, ln2_g, ln2_b, fc2_w, fc2_b,
      ln0_g, ln0_b, gate_w, gate_b, ln3_g, ln3_b, fc3_w, fc3_b,
      ln4_g, ln4_b, fc4_w, fc4_b, (float*)d_out);
}
